// MultiHeadAttention_36644660969882
// MI455X (gfx1250) — matrix operand order repaired and checked by reading, not compiled, not run on hardware
//
#include <hip/hip_runtime.h>

#define HD __device__ __forceinline__

typedef __attribute__((ext_vector_type(16))) __bf16        v16bf;
typedef __attribute__((ext_vector_type(8)))  float         v8f;
typedef __attribute__((ext_vector_type(4)))  unsigned int  u32x4;
typedef __attribute__((ext_vector_type(4)))  float         f32x4;

union FragU { v16bf v; u32x4 q[2]; };

constexpr int kB = 4, kS = 2048, kE = 768, kH = 12, kDH = 64;

HD unsigned short f2bf(float f) {
  unsigned int u = __float_as_uint(f);
  u += 0x7FFFu + ((u >> 16) & 1u);          // round-to-nearest-even
  return (unsigned short)(u >> 16);
}

// Async DMA: global -> LDS, 16 bytes per lane, tracked by ASYNCcnt.
HD void async_b128(const unsigned short* g, unsigned short* lds) {
  unsigned loff = (unsigned)(unsigned long long)lds;  // low 32b = LDS offset
  unsigned long long ga = (unsigned long long)g;
  asm volatile("global_load_async_to_lds_b128 %0, %1, off"
               :: "v"(loff), "v"(ga)
               : "memory");
}
HD void wait_async0() { asm volatile("s_wait_asynccnt 0" ::: "memory"); }
HD void wait_ds0()    { asm volatile("s_wait_dscnt 0" ::: "memory"); }

// A-operand fragment (16x32 bf16): lane holds row M=lane%16.
// element e -> K = (e/8)*16 + (lane/16)*8 + (e%8)  => two contiguous 16B LDS loads.
HD v16bf load_frag_a(const unsigned short* rowbase, int lane) {
  FragU f;
  const int o = ((lane >> 4) << 3);
  f.q[0] = *(const u32x4*)(rowbase + o);
  f.q[1] = *(const u32x4*)(rowbase + 16 + o);
  return f.v;
}

// B-operand fragment (32x16 bf16): lane holds col N=lane%16.
// element e -> K = (e/8)*16 + (lane/16)*8 + (e%8)  => two contiguous 16B LDS loads.
HD v16bf load_frag_b(const unsigned short* colbase, int lane) {
  FragU f;
  const int o = ((lane >> 4) << 3);
  f.q[0] = *(const u32x4*)(colbase + o);
  f.q[1] = *(const u32x4*)(colbase + 16 + o);
  return f.v;
}

HD v8f wmma_bf16(v16bf a, v16bf b, v8f c) {
  return __builtin_amdgcn_wmma_f32_16x16x32_bf16(false, a, false, b, (short)0, c,
                                                 false, false);
}

// ---------------------------------------------------------------------------
// Kernel 1: fused QKV projection. grid = (S*B/64, H, 3).  64x64 tile / block.
// Output bf16 [B,H,S,DH]; Q pre-scaled by 1/sqrt(DH), bias added.
// ---------------------------------------------------------------------------
__global__ __launch_bounds__(128) void qkv_proj_kernel(
    const float* __restrict__ xq, const float* __restrict__ xk,
    const float* __restrict__ xv, const float* __restrict__ wq,
    const float* __restrict__ wk, const float* __restrict__ wv,
    const float* __restrict__ bq, const float* __restrict__ bk,
    const float* __restrict__ bv, unsigned short* __restrict__ Qb,
    unsigned short* __restrict__ Kb, unsigned short* __restrict__ Vb) {
  constexpr int LDA = 40, LDB = 40;
  __shared__ alignas(16) unsigned short sA[64 * LDA];   // [token][k]
  __shared__ alignas(16) unsigned short sB[64 * LDB];   // [n][k] (W^T)

  const int tid = threadIdx.x, lane = tid & 31, wave = tid >> 5;
  const int wm = (wave >> 1) * 32, wn = (wave & 1) * 32;
  const int mt = blockIdx.x, h = blockIdx.y, z = blockIdx.z;

  const float* X  = (z == 0) ? xq : (z == 1) ? xk : xv;
  const float* W  = ((z == 0) ? wq : (z == 1) ? wk : wv) + (size_t)h * kE * kDH;
  const float* Bi = ((z == 0) ? bq : (z == 1) ? bk : bv) + h * kDH;
  unsigned short* Out = (z == 0) ? Qb : (z == 1) ? Kb : Vb;
  const float scale = (z == 0) ? 0.125f : 1.0f;

  v8f acc[2][2] = {};

  for (int k0 = 0; k0 < kE; k0 += 32) {
    __syncthreads();
    // stage A: 64 tokens x 32 k, fp32 -> bf16
#pragma unroll
    for (int i = 0; i < 4; ++i) {
      int f = tid + i * 128;                 // 512 float4
      int row = f >> 3, kc = (f & 7) << 2;
      f32x4 v = *(const f32x4*)(X + (size_t)(mt * 64 + row) * kE + k0 + kc);
      unsigned short* d = &sA[row * LDA + kc];
      d[0] = f2bf(v.x); d[1] = f2bf(v.y); d[2] = f2bf(v.z); d[3] = f2bf(v.w);
    }
    // stage W^T: sB[n][e]
#pragma unroll
    for (int i = 0; i < 4; ++i) {
      int f = tid + i * 128;                 // 512 float4
      int e = f >> 4, n = (f & 15) << 2;
      f32x4 v = *(const f32x4*)(W + (size_t)(k0 + e) * kDH + n);
      sB[(n + 0) * LDB + e] = f2bf(v.x);
      sB[(n + 1) * LDB + e] = f2bf(v.y);
      sB[(n + 2) * LDB + e] = f2bf(v.z);
      sB[(n + 3) * LDB + e] = f2bf(v.w);
    }
    __syncthreads();

    v16bf afrag[2], bfrag[2];
#pragma unroll
    for (int i = 0; i < 2; ++i)
      afrag[i] = load_frag_a(&sA[(wm + i * 16 + (lane & 15)) * LDA], lane);
#pragma unroll
    for (int j = 0; j < 2; ++j)
      bfrag[j] = load_frag_b(&sB[(wn + j * 16 + (lane & 15)) * LDB], lane);
#pragma unroll
    for (int i = 0; i < 2; ++i)
#pragma unroll
      for (int j = 0; j < 2; ++j)
        acc[i][j] = wmma_bf16(afrag[i], bfrag[j], acc[i][j]);
  }

#pragma unroll
  for (int i = 0; i < 2; ++i)
#pragma unroll
    for (int j = 0; j < 2; ++j) {
      const int d = wn + j * 16 + (lane & 15);
      const float bia = Bi[d];
#pragma unroll
      for (int r = 0; r < 8; ++r) {
        int t = mt * 64 + wm + i * 16 + r + ((lane >> 4) << 3);
        int b = t >> 11, s = t & (kS - 1);
        float val = (acc[i][j][r] + bia) * scale;
        Out[(((size_t)b * kH + h) * kS + s) * kDH + d] = f2bf(val);
      }
    }
}

// ---------------------------------------------------------------------------
// Kernel 2: flash attention. grid = (S/64, H, B). 64 q-rows per block,
// each wave owns 16 q-rows; streams K/V in 64-row tiles with online softmax.
// Q/K tiles staged with async DMA (ASYNCcnt); V^T needs transpose -> VGPR path.
// ---------------------------------------------------------------------------
__global__ __launch_bounds__(128) void flash_attn_kernel(
    const unsigned short* __restrict__ Qb, const unsigned short* __restrict__ Kb,
    const unsigned short* __restrict__ Vb, unsigned short* __restrict__ ctx) {
  constexpr int LQ = 40, LK = 40, LV = 72, LP = 72;
  __shared__ alignas(16) unsigned short sQ[64 * LQ];      // [qrow][dh]
  __shared__ alignas(16) unsigned short sK[64 * LK];      // [kvrow][dh]
  __shared__ alignas(16) unsigned short sVT[64 * LV];     // [dh][kv]  (V^T)
  __shared__ alignas(16) unsigned short sP[4][16 * LP];   // per-wave P tile

  const int tid = threadIdx.x, lane = tid & 31, wave = tid >> 5;
  const int qt = blockIdx.x, h = blockIdx.y, b = blockIdx.z;

  const unsigned short* Qh = Qb + (size_t)(b * kH + h) * kS * kDH;
  const unsigned short* Kh = Kb + (size_t)(b * kH + h) * kS * kDH;
  const unsigned short* Vh = Vb + (size_t)(b * kH + h) * kS * kDH;

  // stage Q tile once via async DMA
#pragma unroll
  for (int i = 0; i < 4; ++i) {
    int f = tid + i * 128;                   // 512 x 16B
    int row = f >> 3, c = (f & 7) << 3;
    async_b128(Qh + (size_t)(qt * 64 + row) * kDH + c, &sQ[row * LQ + c]);
  }
  wait_async0();
  __syncthreads();

  v16bf qa[2];
#pragma unroll
  for (int kk = 0; kk < 2; ++kk)
    qa[kk] = load_frag_a(&sQ[(wave * 16 + (lane & 15)) * LQ + kk * 32], lane);

  v8f o[4] = {};
  float m[8], l[8];
#pragma unroll
  for (int r = 0; r < 8; ++r) { m[r] = -1e30f; l[r] = 0.0f; }

  for (int kv0 = 0; kv0 < kS; kv0 += 64) {
    __syncthreads();
    // stage K tile [kv][dh] via async DMA
#pragma unroll
    for (int i = 0; i < 4; ++i) {
      int f = tid + i * 128;
      int row = f >> 3, c = (f & 7) << 3;
      async_b128(Kh + (size_t)(kv0 + row) * kDH + c, &sK[row * LK + c]);
    }
    // stage V^T tile [dh][kv] (transpose -> VGPR path)
#pragma unroll
    for (int i = 0; i < 4; ++i) {
      int f = tid + i * 128;
      int kv = f >> 3, d0 = (f & 7) << 3;
      u32x4 v = *(const u32x4*)(Vh + (size_t)(kv0 + kv) * kDH + d0);
      const unsigned short* pv = (const unsigned short*)&v;
#pragma unroll
      for (int e = 0; e < 8; ++e) sVT[(d0 + e) * LV + kv] = pv[e];
    }
    // prefetch next K/V tiles into cache while this tile computes
    if (kv0 + 64 < kS) {
      __builtin_prefetch(Kh + (size_t)(kv0 + 64 + (tid >> 1)) * kDH, 0, 1);
      __builtin_prefetch(Vh + (size_t)(kv0 + 64 + (tid >> 1)) * kDH, 0, 1);
    }
    wait_async0();
    __syncthreads();

    // S = Q * K^T  (Q pre-scaled by 1/sqrt(DH))
    v8f sf[4] = {};
#pragma unroll
    for (int j = 0; j < 4; ++j)
#pragma unroll
      for (int kk = 0; kk < 2; ++kk) {
        v16bf bk_ = load_frag_b(&sK[(j * 16 + (lane & 15)) * LK + kk * 32], lane);
        sf[j] = wmma_bf16(qa[kk], bk_, sf[j]);
      }

    // online softmax: rows live across 16-lane halves; reduce with shfl_xor.
#pragma unroll
    for (int r = 0; r < 8; ++r) {
      float vmax = fmaxf(fmaxf(sf[0][r], sf[1][r]), fmaxf(sf[2][r], sf[3][r]));
#pragma unroll
      for (int msk = 1; msk < 16; msk <<= 1)
        vmax = fmaxf(vmax, __shfl_xor(vmax, msk, 32));
      float mnew = fmaxf(m[r], vmax);
      float alpha = __expf(m[r] - mnew);
      float rs = 0.0f;
#pragma unroll
      for (int j = 0; j < 4; ++j) {
        float p = __expf(sf[j][r] - mnew);
        sf[j][r] = p;
        rs += p;
      }
#pragma unroll
      for (int msk = 1; msk < 16; msk <<= 1) rs += __shfl_xor(rs, msk, 32);
      l[r] = l[r] * alpha + rs;
      m[r] = mnew;
#pragma unroll
      for (int jj = 0; jj < 4; ++jj) o[jj][r] *= alpha;
      int rowl = r + ((lane >> 4) << 3);
#pragma unroll
      for (int j = 0; j < 4; ++j)
        sP[wave][rowl * LP + j * 16 + (lane & 15)] = f2bf(sf[j][r]);
    }

    // same-wave LDS write -> read (C-layout P -> A-layout reload)
    wait_ds0();

    // O += P * V
#pragma unroll
    for (int jj = 0; jj < 4; ++jj)
#pragma unroll
      for (int kk = 0; kk < 2; ++kk) {
        v16bf pa = load_frag_a(&sP[wave][(lane & 15) * LP + kk * 32], lane);
        v16bf vb = load_frag_b(&sVT[(jj * 16 + (lane & 15)) * LV + kk * 32], lane);
        o[jj] = wmma_bf16(pa, vb, o[jj]);
      }
  }

  // normalize and store ctx as bf16 [B,S,H*DH] (concat-heads layout)
#pragma unroll
  for (int r = 0; r < 8; ++r) {
    float inv = 1.0f / l[r];
    int s = qt * 64 + wave * 16 + r + ((lane >> 4) << 3);
#pragma unroll
    for (int jj = 0; jj < 4; ++jj) {
      float val = o[jj][r] * inv;
      ctx[((size_t)b * kS + s) * kE + h * kDH + jj * 16 + (lane & 15)] =
          f2bf(val);
    }
  }
}

// ---------------------------------------------------------------------------
// Kernel 3: output projection ctx(bf16) @ wo + bo -> fp32. grid = (128, 12).
// ctx tile staged with async DMA; wo needs fp32->bf16 convert -> VGPR path.
// ---------------------------------------------------------------------------
__global__ __launch_bounds__(128) void out_proj_kernel(
    const unsigned short* __restrict__ ctx, const float* __restrict__ wo,
    const float* __restrict__ bo, float* __restrict__ out) {
  constexpr int LDA = 40, LDB = 40;
  __shared__ alignas(16) unsigned short sA[64 * LDA];
  __shared__ alignas(16) unsigned short sB[64 * LDB];

  const int tid = threadIdx.x, lane = tid & 31, wave = tid >> 5;
  const int wm = (wave >> 1) * 32, wn = (wave & 1) * 32;
  const int mt = blockIdx.x, nt = blockIdx.y;

  v8f acc[2][2] = {};

  for (int k0 = 0; k0 < kE; k0 += 32) {
    __syncthreads();
    // stage A (already bf16) via async DMA
#pragma unroll
    for (int i = 0; i < 2; ++i) {
      int f = tid + i * 128;                 // 256 x 16B
      int row = f >> 2, c = (f & 3) << 3;
      async_b128(ctx + (size_t)(mt * 64 + row) * kE + k0 + c,
                 &sA[row * LDA + c]);
    }
    // stage wo^T: sB[n][e], fp32 -> bf16
#pragma unroll
    for (int i = 0; i < 4; ++i) {
      int f = tid + i * 128;
      int e = f >> 4, n = (f & 15) << 2;
      f32x4 v = *(const f32x4*)(wo + (size_t)(k0 + e) * kE + nt * 64 + n);
      sB[(n + 0) * LDB + e] = f2bf(v.x);
      sB[(n + 1) * LDB + e] = f2bf(v.y);
      sB[(n + 2) * LDB + e] = f2bf(v.z);
      sB[(n + 3) * LDB + e] = f2bf(v.w);
    }
    wait_async0();
    __syncthreads();

    v16bf afrag[2], bfrag[2];
#pragma unroll
    for (int i = 0; i < 2; ++i)
      afrag[i] = load_frag_a(&sA[(wm + i * 16 + (lane & 15)) * LDA], lane);
#pragma unroll
    for (int j = 0; j < 2; ++j)
      bfrag[j] = load_frag_b(&sB[(wn + j * 16 + (lane & 15)) * LDB], lane);
#pragma unroll
    for (int i = 0; i < 2; ++i)
#pragma unroll
      for (int j = 0; j < 2; ++j)
        acc[i][j] = wmma_bf16(afrag[i], bfrag[j], acc[i][j]);
  }

#pragma unroll
  for (int i = 0; i < 2; ++i)
#pragma unroll
    for (int j = 0; j < 2; ++j) {
      int n = nt * 64 + wn + j * 16 + (lane & 15);
      float bia = bo[n];
#pragma unroll
      for (int r = 0; r < 8; ++r) {
        int t = mt * 64 + wm + i * 16 + r + ((lane >> 4) << 3);
        out[(size_t)t * kE + n] = acc[i][j][r] + bia;
      }
    }
}

// ---------------------------------------------------------------------------
extern "C" void kernel_launch(void* const* d_in, const int* in_sizes, int n_in,
                              void* d_out, int out_size, void* d_ws,
                              size_t ws_size, hipStream_t stream) {
  (void)in_sizes; (void)n_in; (void)out_size; (void)ws_size;
  const float* query = (const float*)d_in[0];
  const float* key_  = (const float*)d_in[1];
  const float* value = (const float*)d_in[2];
  const float* wq = (const float*)d_in[3];
  const float* bq = (const float*)d_in[4];
  const float* wk = (const float*)d_in[5];
  const float* bk = (const float*)d_in[6];
  const float* wv = (const float*)d_in[7];
  const float* bv = (const float*)d_in[8];
  const float* wo = (const float*)d_in[9];
  const float* bo = (const float*)d_in[10];
  float* out = (float*)d_out;

  unsigned short* ws = (unsigned short*)d_ws;
  const size_t qkvElems = (size_t)kB * kH * kS * kDH;   // 6,291,456
  unsigned short* Qb  = ws;
  unsigned short* Kb  = Qb + qkvElems;
  unsigned short* Vb  = Kb + qkvElems;
  unsigned short* ctx = Vb + qkvElems;                  // [B,S,E] bf16

  dim3 g1(kB * kS / 64, kH, 3);
  qkv_proj_kernel<<<g1, 128, 0, stream>>>(query, key_, value, wq, wk, wv, bq,
                                          bk, bv, Qb, Kb, Vb);
  dim3 g2(kS / 64, kH, kB);
  flash_attn_kernel<<<g2, 128, 0, stream>>>(Qb, Kb, Vb, ctx);
  dim3 g3(kB * kS / 64, kE / 64, 1);
  out_proj_kernel<<<g3, 128, 0, stream>>>(ctx, wo, bo, out);
}
